// SpKBGATModified_81552839016626
// MI455X (gfx1250) — matrix operand order repaired and checked by reading, not compiled, not run on hardware
//
#include <hip/hip_runtime.h>
#include <hip/hip_bf16.h>
#include <math.h>

// ---------------------------------------------------------------------------
// Types for CDNA5 WMMA (gfx1250, wave32)
// ---------------------------------------------------------------------------
typedef _Float16 h8   __attribute__((ext_vector_type(8)));
typedef _Float16 v16h __attribute__((ext_vector_type(16)));
typedef float    v8f  __attribute__((ext_vector_type(8)));

#define N_NODES    40000
#define N_REL      474
#define F_ENT      100     // ENT_DIM == REL_DIM == NHID
#define F_OUT      200     // OUT_DIM = NHID*NHEADS
#define E_BASE     100000
#define E_NHOP     30000
#define E_TOT      130000
#define N_BATCH    8192
#define EPSF       1e-12f
#define LRELU_A    0.2f
#define OLAMBDA    0.01f

// ---------------------------------------------------------------------------
// 1) row L2-normalize (f32 in) -> zero-padded f16 row (Kp stride)
// ---------------------------------------------------------------------------
__global__ void k_l2norm_f16(const float* __restrict__ src, _Float16* __restrict__ dst,
                             int M, int C, int Kp) {
  int wave = (int)((blockIdx.x * blockDim.x + threadIdx.x) >> 5);
  int lane = threadIdx.x & 31;
  if (wave >= M) return;
  const float* row = src + (size_t)wave * C;
  float ss = 0.f;
  for (int j = lane; j < C; j += 32) { float v = row[j]; ss += v * v; }
  for (int off = 16; off; off >>= 1) ss += __shfl_xor(ss, off, 32);
  float inv = 1.f / fmaxf(sqrtf(ss), EPSF);
  _Float16* drow = dst + (size_t)wave * Kp;
  for (int j = lane; j < Kp; j += 32)
    drow[j] = (j < C) ? (_Float16)(row[j] * inv) : (_Float16)0.f;
}

// ---------------------------------------------------------------------------
// 2) f32 -> f16 with K zero-padding
// ---------------------------------------------------------------------------
__global__ void k_cvt_pad(const float* __restrict__ src, _Float16* __restrict__ dst,
                          int M, int C, int Kp) {
  size_t id = (size_t)blockIdx.x * blockDim.x + threadIdx.x;
  size_t total = (size_t)M * Kp;
  if (id >= total) return;
  int k = (int)(id % Kp);
  size_t m = id / Kp;
  dst[id] = (k < C) ? (_Float16)src[m * C + k] : (_Float16)0.f;
}

// ---------------------------------------------------------------------------
// 3) Pack B (KxN logical, from f32 source, optional transpose+col offset) into
//    WMMA B-operand order: tile(kt,nt) -> lane -> 16 halves, where
//    lane = n%16 + 16*g ; element j corresponds to K = kt*32 + 16*g + j.
// ---------------------------------------------------------------------------
__global__ void k_pack_b(const float* __restrict__ src, int ld, int coff,
                         int Klim, int Nlim, int transpose,
                         _Float16* __restrict__ dst, int ktiles, int ntiles) {
  int total = ktiles * ntiles * 512;
  int id = blockIdx.x * blockDim.x + threadIdx.x;
  if (id >= total) return;
  int j    = id & 15;
  int lane = (id >> 4) & 31;
  int tile = id >> 9;
  int nt = tile % ntiles, kt = tile / ntiles;
  int n = nt * 16 + (lane & 15);
  int g = lane >> 4;
  int k = kt * 32 + g * 16 + j;
  float v = 0.f;
  if (k < Klim && n < Nlim)
    v = transpose ? src[(size_t)n * ld + k + coff]
                  : src[(size_t)k * ld + n + coff];
  dst[id] = (_Float16)v;
}

// ---------------------------------------------------------------------------
// 4) WMMA GEMM, templated on padded K (full unroll: 4 or 7 chained
//    v_wmma_f32_16x16x32_f16) and on N (constant ldc -> constant-offset
//    global_store_b32 in the epilogue). One wave per 16x16 C tile.
//    A guard replaced by row-clamp; wave-uniform full-tile fast path for
//    the C store (boundary tiles take the per-lane predicated path).
//    A-operand in contiguous K order (matches packed B): lane (m=l%16, g=l/16) reads the
//    two contiguous 8-half chunks at K offsets 16g and 16g+8 (b128 each).
// ---------------------------------------------------------------------------
template <int KP, int NN>
__global__ void k_wmma_gemm(const _Float16* __restrict__ A,
                            const _Float16* __restrict__ Bp,
                            float* __restrict__ C,
                            int M) {
  constexpr int KSTEPS = KP / 32;
  constexpr int NTILES = (NN + 15) / 16;
  constexpr size_t BSTRIDE = (size_t)NTILES * 512;  // halves per k-tile of packed B
  int mtiles = (M + 15) >> 4;
  int tile = (int)((blockIdx.x * blockDim.x + threadIdx.x) >> 5);
  if (tile >= mtiles * NTILES) return;
  int mt = tile / NTILES, nt = tile - mt * NTILES;
  int lane = threadIdx.x & 31;
  int lmod = lane & 15, g = lane >> 4;
  int m = mt * 16 + lmod;
  int mc = (m < M) ? m : (M - 1);          // clamp instead of predicate

  const _Float16* ap = A + (size_t)mc * KP + g * 16;
  const _Float16* bp = Bp + ((size_t)nt * 32 + lane) * 16;

  v8f acc = {0.f, 0.f, 0.f, 0.f, 0.f, 0.f, 0.f, 0.f};
#pragma unroll
  for (int kt = 0; kt < KSTEPS; ++kt) {
    h8 lo = *(const h8*)(ap + kt * 32);
    h8 hi = *(const h8*)(ap + kt * 32 + 8);
    v16h a;
#pragma unroll
    for (int i = 0; i < 8; ++i) { a[i] = lo[i]; a[8 + i] = hi[i]; }
    v16h b = *(const v16h*)(bp + kt * BSTRIDE);
    acc = __builtin_amdgcn_wmma_f32_16x16x32_f16(false, a, false, b,
                                                 (short)0, acc, false, false);
  }

  // D layout: vgpr r: lanes 0-15 -> M=r, lanes 16-31 -> M=r+8 (N = lane%16)
  int n = nt * 16 + lmod;
  float* cp = C + (size_t)(mt * 16 + g * 8) * NN + n;
  bool full = (mt * 16 + 16 <= M) && (nt * 16 + 16 <= NN);  // wave-uniform
  if (full) {
#pragma unroll
    for (int r = 0; r < 8; ++r) cp[(size_t)r * NN] = acc[r];
  } else {
#pragma unroll
    for (int r = 0; r < 8; ++r) {
      int row = mt * 16 + r + g * 8;
      if (row < M && n < NN) cp[(size_t)r * NN] = acc[r];
    }
  }
}

// ---------------------------------------------------------------------------
// 5) Edge attention (fused): edge_m = Tt[tgt]+Ts[src]+Rr[...]; softmax numer /
//    denom via atomics. One wave per edge.
// ---------------------------------------------------------------------------
template <int NOUT>
__global__ void k_edge_att(const float* __restrict__ Tt, const float* __restrict__ Ts,
                           const float* __restrict__ Rr, const float* __restrict__ a2,
                           float* __restrict__ h, float* __restrict__ rs,
                           const int* __restrict__ edge_list,
                           const int* __restrict__ edge_type,
                           const int* __restrict__ nhop) {
  constexpr int R = (NOUT + 31) / 32;
  int e = (int)((blockIdx.x * blockDim.x + threadIdx.x) >> 5);
  int lane = threadIdx.x & 31;
  if (e >= E_TOT) return;

  int t, s;
  float rrow[R];
  if (e < E_BASE) {
    t = edge_list[e];
    s = edge_list[E_BASE + e];
    const float* rp = Rr + (size_t)edge_type[e] * NOUT;
#pragma unroll
    for (int r = 0; r < R; ++r) { int j = lane + 32 * r; rrow[r] = (j < NOUT) ? rp[j] : 0.f; }
  } else {
    int k = e - E_BASE;
    t = nhop[k * 4 + 3];
    s = nhop[k * 4 + 0];
    const float* rp1 = Rr + (size_t)nhop[k * 4 + 1] * NOUT;
    const float* rp2 = Rr + (size_t)nhop[k * 4 + 2] * NOUT;
#pragma unroll
    for (int r = 0; r < R; ++r) { int j = lane + 32 * r; rrow[r] = (j < NOUT) ? (rp1[j] + rp2[j]) : 0.f; }
  }
  const float* tp = Tt + (size_t)t * NOUT;
  const float* sp = Ts + (size_t)s * NOUT;
  float m[R];
  float dot = 0.f;
#pragma unroll
  for (int r = 0; r < R; ++r) {
    int j = lane + 32 * r;
    float v = 0.f;
    if (j < NOUT) { v = tp[j] + sp[j] + rrow[r]; dot += v * a2[j]; }
    m[r] = v;
  }
  for (int off = 16; off; off >>= 1) dot += __shfl_xor(dot, off, 32);
  float lr = (dot >= 0.f) ? dot : LRELU_A * dot;
  float ee = __expf(-lr);
  if (lane == 0) atomicAdd(&rs[t], ee);
#pragma unroll
  for (int r = 0; r < R; ++r) {
    int j = lane + 32 * r;
    if (j < NOUT) atomicAdd(&h[(size_t)t * NOUT + j], ee * m[r]);
  }
}

// ---------------------------------------------------------------------------
// 6) Layer-1 finalize: x[:, head*100+jj] = elu(h_head / rowsum_head)
// ---------------------------------------------------------------------------
__global__ void k_finalize1(const float* __restrict__ h0, const float* __restrict__ h1,
                            const float* __restrict__ rs0, const float* __restrict__ rs1,
                            float* __restrict__ x) {
  int idx = blockIdx.x * blockDim.x + threadIdx.x;
  if (idx >= N_NODES * F_OUT) return;
  int n = idx / F_OUT, j = idx - n * F_OUT;
  const float* h  = (j < F_ENT) ? h0 : h1;
  const float* rs = (j < F_ENT) ? rs0 : rs1;
  int jj = (j < F_ENT) ? j : (j - F_ENT);
  float d = rs[n]; d = (d == 0.f) ? EPSF : d;
  float v = h[(size_t)n * F_ENT + jj] / d;
  v = (v >= 0.f) ? v : (__expf(v) - 1.f);   // ELU
  x[idx] = v;
}

// ---------------------------------------------------------------------------
// 7) mask scatter
// ---------------------------------------------------------------------------
__global__ void k_mask(const int* __restrict__ batch, float* __restrict__ mask) {
  int b = blockIdx.x * blockDim.x + threadIdx.x;
  if (b < N_BATCH) mask[batch[b * 3 + 2]] = 1.f;
}

// ---------------------------------------------------------------------------
// 8) out_entity = l2norm(ent@W_ent + mask * (h2/rowsum2)), wave per row
// ---------------------------------------------------------------------------
__global__ void k_out_entity(const float* __restrict__ EW, const float* __restrict__ h2,
                             const float* __restrict__ rs2, const float* __restrict__ mask,
                             float* __restrict__ out) {
  int n = (int)((blockIdx.x * blockDim.x + threadIdx.x) >> 5);
  int lane = threadIdx.x & 31;
  if (n >= N_NODES) return;
  float d = rs2[n]; d = (d == 0.f) ? EPSF : d;
  float mk = mask[n];
  constexpr int R = (F_OUT + 31) / 32;
  float t[R];
  float ss = 0.f;
#pragma unroll
  for (int r = 0; r < R; ++r) {
    int j = lane + 32 * r;
    float v = 0.f;
    if (j < F_OUT) {
      v = EW[(size_t)n * F_OUT + j] + mk * (h2[(size_t)n * F_OUT + j] / d);
      ss += v * v;
    }
    t[r] = v;
  }
  for (int off = 16; off; off >>= 1) ss += __shfl_xor(ss, off, 32);
  float inv = 1.f / fmaxf(sqrtf(ss), EPSF);
#pragma unroll
  for (int r = 0; r < R; ++r) {
    int j = lane + 32 * r;
    if (j < F_OUT) out[(size_t)n * F_OUT + j] = t[r] * inv;
  }
}

// ---------------------------------------------------------------------------
// 9) ortho losses: 3 matrices reshaped (2, L); loss = lam*sum((G-I)^2)
// ---------------------------------------------------------------------------
__global__ void k_ortho(const float* __restrict__ att_a, const float* __restrict__ out_a,
                        float* __restrict__ dst) {
  const float* base;
  int L;
  if (blockIdx.x < 2) { base = att_a + (size_t)blockIdx.x * 30000; L = 15000; }
  else                { base = out_a; L = 60000; }
  const float* r0 = base;
  const float* r1 = base + L;
  float d00 = 0.f, d01 = 0.f, d11 = 0.f;
  for (int i = threadIdx.x; i < L; i += blockDim.x) {
    float a = r0[i], b = r1[i];
    d00 += a * a; d01 += a * b; d11 += b * b;
  }
  __shared__ float s0[256], s1[256], s2[256];
  s0[threadIdx.x] = d00; s1[threadIdx.x] = d01; s2[threadIdx.x] = d11;
  __syncthreads();
  for (int off = 128; off; off >>= 1) {
    if ((int)threadIdx.x < off) {
      s0[threadIdx.x] += s0[threadIdx.x + off];
      s1[threadIdx.x] += s1[threadIdx.x + off];
      s2[threadIdx.x] += s2[threadIdx.x + off];
    }
    __syncthreads();
  }
  if (threadIdx.x == 0) {
    float n0 = fmaxf(sqrtf(s0[0]), EPSF), n1 = fmaxf(sqrtf(s2[0]), EPSF);
    float g00 = s0[0] / (n0 * n0), g11 = s2[0] / (n1 * n1), g01 = s1[0] / (n0 * n1);
    float e0 = g00 - 1.f, e1 = g11 - 1.f;
    atomicAdd(dst, OLAMBDA * (e0 * e0 + e1 * e1 + 2.f * g01 * g01));
  }
}

// ---------------------------------------------------------------------------
// host launch
// ---------------------------------------------------------------------------
extern "C" void kernel_launch(void* const* d_in, const int* in_sizes, int n_in,
                              void* d_out, int out_size, void* d_ws, size_t ws_size,
                              hipStream_t stream) {
  (void)in_sizes; (void)n_in; (void)out_size; (void)ws_size;
  const int*   edge_list = (const int*)d_in[0];
  const int*   edge_type = (const int*)d_in[1];
  const int*   batch     = (const int*)d_in[2];
  const int*   nhop      = (const int*)d_in[3];
  const float* ent_emb   = (const float*)d_in[4];
  const float* rel_emb   = (const float*)d_in[5];
  const float* W_ent     = (const float*)d_in[6];
  const float* W_rel     = (const float*)d_in[7];
  const float* att_a     = (const float*)d_in[8];
  const float* att_a2    = (const float*)d_in[9];
  const float* out_a     = (const float*)d_in[10];
  const float* out_a2    = (const float*)d_in[11];

  float* out_entity = (float*)d_out;                       // 40000*200
  float* out_rel1   = (float*)d_out + 8000000;             // 474*200
  float* out_ortho  = (float*)d_out + 8094800;             // scalar

  // bump allocator over workspace
  char* ws = (char*)d_ws;
  size_t off = 0;
  auto alloc = [&](size_t bytes) -> void* {
    void* p = ws + off;
    off = (off + bytes + 255) & ~(size_t)255;
    return p;
  };

  const int KP1 = 128;  // padded K for 100-dim features
  const int KP2 = 224;  // padded K for 200-dim features

  _Float16* ent_h    = (_Float16*)alloc((size_t)N_NODES * KP1 * 2);
  _Float16* rel_h    = (_Float16*)alloc((size_t)N_REL   * KP1 * 2);
  _Float16* outrel_h = (_Float16*)alloc((size_t)N_REL   * KP2 * 2);
  float*    x        = (float*)   alloc((size_t)N_NODES * F_OUT * 4);
  _Float16* x_h      = (_Float16*)alloc((size_t)N_NODES * KP2 * 2);
  float*    Tt0 = (float*)alloc((size_t)N_NODES * F_ENT * 4);
  float*    Ts0 = (float*)alloc((size_t)N_NODES * F_ENT * 4);
  float*    Tt1 = (float*)alloc((size_t)N_NODES * F_ENT * 4);
  float*    Ts1 = (float*)alloc((size_t)N_NODES * F_ENT * 4);
  float*    Rr0 = (float*)alloc((size_t)N_REL * F_ENT * 4);
  float*    Rr1 = (float*)alloc((size_t)N_REL * F_ENT * 4);
  float*    h0  = (float*)alloc((size_t)N_NODES * F_ENT * 4);
  float*    h1  = (float*)alloc((size_t)N_NODES * F_ENT * 4);
  float*    rs0 = (float*)alloc((size_t)N_NODES * 4);
  float*    rs1 = (float*)alloc((size_t)N_NODES * 4);
  float*    Tt2 = (float*)alloc((size_t)N_NODES * F_OUT * 4);
  float*    Ts2 = (float*)alloc((size_t)N_NODES * F_OUT * 4);
  float*    Rr2 = (float*)alloc((size_t)N_REL * F_OUT * 4);
  float*    h2  = (float*)alloc((size_t)N_NODES * F_OUT * 4);
  float*    rs2 = (float*)alloc((size_t)N_NODES * 4);
  float*    EW  = (float*)alloc((size_t)N_NODES * F_OUT * 4);
  float*    msk = (float*)alloc((size_t)N_NODES * 4);

  // packed weight buffers (ktiles*ntiles*512 halves each)
  const int KT1 = KP1 / 32, KT2 = KP2 / 32;     // 4, 7
  const int NT1 = 7, NT2 = 13;                  // ceil(100/16), ceil(200/16)
  _Float16* Bwrel = (_Float16*)alloc((size_t)KT1 * NT2 * 512 * 2);
  _Float16* Bwent = (_Float16*)alloc((size_t)KT1 * NT2 * 512 * 2);
  _Float16* Batt[2][3];
  for (int i = 0; i < 2; ++i)
    for (int p = 0; p < 3; ++p)
      Batt[i][p] = (_Float16*)alloc((size_t)KT1 * NT1 * 512 * 2);
  _Float16* Bout[3];
  for (int p = 0; p < 3; ++p)
    Bout[p] = (_Float16*)alloc((size_t)KT2 * NT2 * 512 * 2);

  // ---- zero-init accumulators / outputs needing init
  hipMemsetAsync(h0,  0, (size_t)N_NODES * F_ENT * 4, stream);
  hipMemsetAsync(h1,  0, (size_t)N_NODES * F_ENT * 4, stream);
  hipMemsetAsync(h2,  0, (size_t)N_NODES * F_OUT * 4, stream);
  hipMemsetAsync(rs0, 0, (size_t)N_NODES * 4, stream);
  hipMemsetAsync(rs1, 0, (size_t)N_NODES * 4, stream);
  hipMemsetAsync(rs2, 0, (size_t)N_NODES * 4, stream);
  hipMemsetAsync(msk, 0, (size_t)N_NODES * 4, stream);
  hipMemsetAsync(out_ortho, 0, 4, stream);

  // ---- normalize entities -> f16 padded
  k_l2norm_f16<<<(N_NODES * 32 + 255) / 256, 256, 0, stream>>>(ent_emb, ent_h,
                                                               N_NODES, F_ENT, KP1);
  // ---- rel embeddings -> f16 padded
  {
    size_t tot = (size_t)N_REL * KP1;
    k_cvt_pad<<<(int)((tot + 255) / 256), 256, 0, stream>>>(rel_emb, rel_h, N_REL, F_ENT, KP1);
  }

  // ---- pack weights
  auto pack = [&](const float* src, int ld, int coff, int Klim, int Nlim, int tr,
                  _Float16* dst, int kt, int nt) {
    int total = kt * nt * 512;
    k_pack_b<<<(total + 255) / 256, 256, 0, stream>>>(src, ld, coff, Klim, Nlim, tr,
                                                      dst, kt, nt);
  };
  pack(W_rel, F_OUT, 0, F_ENT, F_OUT, 0, Bwrel, KT1, NT2);    // B[k][n]=W_rel[k][n]
  pack(W_ent, F_OUT, 0, F_ENT, F_OUT, 0, Bwent, KT1, NT2);
  for (int i = 0; i < 2; ++i)
    for (int p = 0; p < 3; ++p)   // B[k][n]=att_a[i][n][p*100+k]
      pack(att_a + (size_t)i * 100 * 300, 300, p * 100, F_ENT, F_ENT, 1,
           Batt[i][p], KT1, NT1);
  for (int p = 0; p < 3; ++p)     // B[k][n]=out_a[n][p*200+k]
    pack(out_a, 600, p * 200, F_OUT, F_OUT, 1, Bout[p], KT2, NT2);

  // GEMM launchers: KP/N compile-time (128|224, 100|200), M runtime
  auto gemm1_n100 = [&](const _Float16* A, const _Float16* Bp, float* C, int M) {
    int tiles = ((M + 15) / 16) * 7;
    k_wmma_gemm<128, 100><<<(tiles * 32 + 255) / 256, 256, 0, stream>>>(A, Bp, C, M);
  };
  auto gemm1_n200 = [&](const _Float16* A, const _Float16* Bp, float* C, int M) {
    int tiles = ((M + 15) / 16) * 13;
    k_wmma_gemm<128, 200><<<(tiles * 32 + 255) / 256, 256, 0, stream>>>(A, Bp, C, M);
  };
  auto gemm2_n200 = [&](const _Float16* A, const _Float16* Bp, float* C, int M) {
    int tiles = ((M + 15) / 16) * 13;
    k_wmma_gemm<224, 200><<<(tiles * 32 + 255) / 256, 256, 0, stream>>>(A, Bp, C, M);
  };

  // ---- out_relation_1 = rel_emb @ W_rel  (written straight to d_out)
  gemm1_n200(rel_h, Bwrel, out_rel1, N_REL);
  {
    size_t tot = (size_t)N_REL * KP2;
    k_cvt_pad<<<(int)((tot + 255) / 256), 256, 0, stream>>>(out_rel1, outrel_h,
                                                            N_REL, F_OUT, KP2);
  }

  // ---- layer-1 per-node / per-relation transforms (WMMA)
  gemm1_n100(ent_h, Batt[0][0], Tt0, N_NODES);
  gemm1_n100(ent_h, Batt[0][1], Ts0, N_NODES);
  gemm1_n100(rel_h, Batt[0][2], Rr0, N_REL);
  gemm1_n100(ent_h, Batt[1][0], Tt1, N_NODES);
  gemm1_n100(ent_h, Batt[1][1], Ts1, N_NODES);
  gemm1_n100(rel_h, Batt[1][2], Rr1, N_REL);

  // ---- layer-1 edge phase (both heads)
  const int eblocks = (E_TOT * 32 + 255) / 256;
  k_edge_att<F_ENT><<<eblocks, 256, 0, stream>>>(Tt0, Ts0, Rr0, att_a2, h0, rs0,
                                                 edge_list, edge_type, nhop);
  k_edge_att<F_ENT><<<eblocks, 256, 0, stream>>>(Tt1, Ts1, Rr1, att_a2 + F_ENT, h1, rs1,
                                                 edge_list, edge_type, nhop);

  // ---- x = concat(elu(h0/rs0), elu(h1/rs1)); convert to f16
  k_finalize1<<<(N_NODES * F_OUT + 255) / 256, 256, 0, stream>>>(h0, h1, rs0, rs1, x);
  {
    size_t tot = (size_t)N_NODES * KP2;
    k_cvt_pad<<<(int)((tot + 255) / 256), 256, 0, stream>>>(x, x_h, N_NODES, F_OUT, KP2);
  }

  // ---- layer-2 transforms (WMMA)
  gemm2_n200(x_h,      Bout[0], Tt2, N_NODES);
  gemm2_n200(x_h,      Bout[1], Ts2, N_NODES);
  gemm2_n200(outrel_h, Bout[2], Rr2, N_REL);

  // ---- layer-2 edge phase
  k_edge_att<F_OUT><<<eblocks, 256, 0, stream>>>(Tt2, Ts2, Rr2, out_a2, h2, rs2,
                                                 edge_list, edge_type, nhop);

  // ---- ent @ W_entities (WMMA), mask, final combine + l2norm
  gemm1_n200(ent_h, Bwent, EW, N_NODES);
  k_mask<<<(N_BATCH + 255) / 256, 256, 0, stream>>>(batch, msk);
  k_out_entity<<<(N_NODES * 32 + 255) / 256, 256, 0, stream>>>(EW, h2, rs2, msk,
                                                               out_entity);

  // ---- ortho losses
  k_ortho<<<3, 256, 0, stream>>>(att_a, out_a, out_ortho);
}
